// GAT_student_11003706212773
// MI455X (gfx1250) — compile-verified, matrix operand order checked by reading, not run on hardware
//
#include <hip/hip_runtime.h>
#include <hip/hip_bf16.h>

typedef __attribute__((ext_vector_type(16))) __bf16 v16bf;
typedef __attribute__((ext_vector_type(8)))  float  v8f;

#define NEG_SLOPE 0.2f

// ---------------------------------------------------------------- utilities
__global__ void k_cvt_bf16(__bf16* __restrict__ o, const float* __restrict__ in, long n) {
  long t = (long)blockIdx.x * blockDim.x + threadIdx.x;
  if (t < n) o[t] = (__bf16)in[t];
}

// oT[n*K + k] = (bf16) in[k*Nout + n]   (in is [K, Nout] row-major)
__global__ void k_transpose_bf16(__bf16* __restrict__ oT, const float* __restrict__ in,
                                 int K, int Nout) {
  long t = (long)blockIdx.x * blockDim.x + threadIdx.x;
  if (t >= (long)K * Nout) return;
  int k = (int)(t / Nout), n = (int)(t % Nout);
  oT[(long)n * K + k] = (__bf16)in[t];
}

__global__ void k_fill_f32(float* p, float v, long n) {
  long t = (long)blockIdx.x * blockDim.x + threadIdx.x;
  if (t < n) p[t] = v;
}
__global__ void k_fill_i32(int* p, int v, long n) {
  long t = (long)blockIdx.x * blockDim.x + threadIdx.x;
  if (t < n) p[t] = v;
}

// ---------------------------------------------------------------- WMMA GEMM
// C[M,Nout] (f32, row-major) = A[M,K] (bf16, row-major) * Bt[Nout,K]^T (bf16)
// block = (32,4): 4 waves, each computing one 16x16 tile.
// grid = (M/16, Nout/64). K % 32 == 0, M % 16 == 0, Nout % 64 == 0.
__global__ __launch_bounds__(128)
void k_gemm_bf16_wmma(const __bf16* __restrict__ A, const __bf16* __restrict__ Bt,
                      float* __restrict__ C, int Nout, int K) {
  const int lane  = threadIdx.x;                 // 0..31
  const int wave  = threadIdx.y;                 // 0..3
  const int m0    = blockIdx.x * 16;
  const int n0    = (blockIdx.y * 4 + wave) * 16;
  const int rowA  = m0 + (lane & 15);
  const int coln  = n0 + (lane & 15);
  const int khalf = (lane >= 16) ? 8 : 0;        // 16-bit A/B fragment lane split

  v8f acc = {};
  for (int kb = 0; kb < K; kb += 32) {
    union Frag { v16bf v; unsigned u[8]; } a, b;
#pragma unroll
    for (int p = 0; p < 8; ++p) {
      // K index pattern for 16-bit A fragment (ISA 7.12.2):
      // v0..3 -> K = {0..7|8..15}, v4..7 -> K = {16..23|24..31}
      const int k = kb + ((p >= 4) ? 16 : 0) + khalf + 2 * (p & 3);
      a.u[p] = *(const unsigned*)(A  + (long)rowA * K + k);
      b.u[p] = *(const unsigned*)(Bt + (long)coln * K + k);
    }
    acc = __builtin_amdgcn_wmma_f32_16x16x32_bf16(
        /*neg_a=*/false, a.v, /*neg_b=*/false, b.v,
        /*c_mod=*/(short)0, acc, /*reuse_a=*/false, /*reuse_b=*/false);
  }
#pragma unroll
  for (int j = 0; j < 8; ++j) {
    const int row = m0 + ((lane < 16) ? j : (8 + j));
    C[(long)row * Nout + coln] = acc[j];
  }
}

// ---------------------------------------------------------------- attention
// el[n,h] = sum_f feat[n,h,f]*al[h,f];  er likewise
__global__ void k_dots(const float* __restrict__ feat, const float* __restrict__ al,
                       const float* __restrict__ ar, float* __restrict__ el,
                       float* __restrict__ er, int n_nodes, int H, int F) {
  long t = (long)blockIdx.x * blockDim.x + threadIdx.x;
  if (t >= (long)n_nodes * H) return;
  int n = (int)(t / H), h = (int)(t % H);
  const float* fp  = feat + ((long)n * H + h) * F;
  const float* alp = al + (long)h * F;
  const float* arp = ar + (long)h * F;
  float sl = 0.f, sr = 0.f;
  for (int f = 0; f < F; ++f) { float v = fp[f]; sl += v * alp[f]; sr += v * arp[f]; }
  el[t] = sl; er[t] = sr;
}

__device__ __forceinline__ int enc_ord(float f) {
  int o = __float_as_int(f);
  return (o >= 0) ? o : (o ^ 0x7fffffff);
}
__device__ __forceinline__ float dec_ord(int o) {
  if (o == (int)0x80000000) return 0.0f;   // no in-edges: reference maps -inf -> 0
  return (o >= 0) ? __int_as_float(o) : __int_as_float(o ^ 0x7fffffff);
}

__global__ void k_edge_e(const int* __restrict__ src, const int* __restrict__ dst,
                         const float* __restrict__ el, const float* __restrict__ er,
                         float* __restrict__ ebuf, int* __restrict__ mbuf, int E, int H) {
  long t = (long)blockIdx.x * blockDim.x + threadIdx.x;
  if (t >= (long)E * H) return;
  int h = (int)(t % H), e = (int)(t / H);
  float v = el[(long)src[e] * H + h] + er[(long)dst[e] * H + h];
  v = (v > 0.f) ? v : NEG_SLOPE * v;     // leaky_relu
  ebuf[t] = v;
  atomicMax(&mbuf[(long)dst[e] * H + h], enc_ord(v));
}

__global__ void k_edge_exp(const int* __restrict__ dst, float* __restrict__ ebuf,
                           const int* __restrict__ mbuf, float* __restrict__ den,
                           int E, int H) {
  long t = (long)blockIdx.x * blockDim.x + threadIdx.x;
  if (t >= (long)E * H) return;
  int h = (int)(t % H), e = (int)(t / H);
  float m  = dec_ord(mbuf[(long)dst[e] * H + h]);
  float ex = expf(ebuf[t] - m);
  ebuf[t] = ex;
  atomicAdd(&den[(long)dst[e] * H + h], ex);
}

__global__ void k_edge_alpha(const int* __restrict__ dst, const float* __restrict__ ebuf,
                             const float* __restrict__ den, float* __restrict__ alpha,
                             int E, int H) {
  long t = (long)blockIdx.x * blockDim.x + threadIdx.x;
  if (t >= (long)E * H) return;
  int h = (int)(t % H), e = (int)(t / H);
  alpha[t] = ebuf[t] / fmaxf(den[(long)dst[e] * H + h], 1e-9f);
}

// scatter-add: rst[dst,h,f] += feat[src,h,f]*alpha[e,h]; one thread = 4 features
__global__ void k_edge_msg(const int* __restrict__ src, const int* __restrict__ dst,
                           const float* __restrict__ alpha, const float* __restrict__ feat,
                           float* __restrict__ rst, int E, int H, int F) {
  const int Fq = F >> 2;
  long t = (long)blockIdx.x * blockDim.x + threadIdx.x;
  if (t >= (long)E * H * Fq) return;
  int f4 = (int)(t % Fq);
  long r = t / Fq;
  int h = (int)(r % H), e = (int)(r / H);
  float a = alpha[(long)e * H + h];
  const float4 v = *(const float4*)(feat + ((long)src[e] * H + h) * F + f4 * 4);
  float* o = rst + ((long)dst[e] * H + h) * F + f4 * 4;
  atomicAdd(o + 0, v.x * a);
  atomicAdd(o + 1, v.y * a);
  atomicAdd(o + 2, v.z * a);
  atomicAdd(o + 3, v.w * a);
}

// h = act(rst + res + b); write f32 and bf16 copies
__global__ void k_post(const float* __restrict__ rst, const float* __restrict__ res,
                       const float* __restrict__ b, float* __restrict__ hf,
                       __bf16* __restrict__ hbf, int HF, int act, long total) {
  long t = (long)blockIdx.x * blockDim.x + threadIdx.x;
  if (t >= total) return;
  float v = rst[t] + (res ? res[t] : 0.f) + b[t % HF];
  if (act) v = (v > 0.f) ? v : expm1f(v);          // elu
  hf[t] = v;
  hbf[t] = (__bf16)v;
}

__global__ void k_prior(const float* __restrict__ hf, float* __restrict__ prior,
                        int n_nodes, int HF) {
  long t = (long)blockIdx.x * blockDim.x + threadIdx.x;
  if (t >= n_nodes) return;
  const float* p = hf + t * (long)HF;
  float s = 0.f;
  for (int j = 0; j < HF; ++j) s += p[j];
  prior[t] = s / (float)HF;
}

// logits[n,c] = mean_h (rst + res + b)[n,h,c]
__global__ void k_final(const float* __restrict__ rst, const float* __restrict__ res,
                        const float* __restrict__ b, float* __restrict__ logits,
                        int n_nodes, int H, int C) {
  long t = (long)blockIdx.x * blockDim.x + threadIdx.x;
  if (t >= (long)n_nodes * C) return;
  int n = (int)(t / C), c = (int)(t % C);
  float s = 0.f;
  for (int h = 0; h < H; ++h) {
    long idx = ((long)n * H + h) * C + c;
    s += rst[idx] + res[idx] + b[h * C + c];
  }
  logits[t] = s / (float)H;
}

// ---------------------------------------------------------------- host
extern "C" void kernel_launch(void* const* d_in, const int* in_sizes, int n_in,
                              void* d_out, int out_size, void* d_ws, size_t ws_size,
                              hipStream_t stream) {
  enum { NN = 20000, EE = 320000, DIN = 512, FH = 128, CC = 64, HH = 4, HF1 = HH * FH, HF2 = HH * CC };

  const float* x     = (const float*)d_in[0];
  const int*   src   = (const int*)  d_in[1];
  const int*   dst   = (const int*)  d_in[2];
  const float* W0    = (const float*)d_in[3];
  const float* al0   = (const float*)d_in[4];
  const float* ar0   = (const float*)d_in[5];
  const float* b0    = (const float*)d_in[6];
  const float* W1    = (const float*)d_in[7];
  const float* al1   = (const float*)d_in[8];
  const float* ar1   = (const float*)d_in[9];
  const float* b1    = (const float*)d_in[10];
  const float* resW1 = (const float*)d_in[11];
  const float* W2    = (const float*)d_in[12];
  const float* al2   = (const float*)d_in[13];
  const float* ar2   = (const float*)d_in[14];
  const float* b2    = (const float*)d_in[15];
  const float* resW2 = (const float*)d_in[16];

  float* out    = (float*)d_out;
  float* logits = out;                    // [N, C]
  float* prior  = out + (long)NN * CC;    // [N]
  float* attout = prior + NN;             // [E, H]

  // workspace carve-up
  char*  w   = (char*)d_ws;
  size_t off = 0;
  auto carve = [&](size_t bytes) -> char* {
    char* p = w + off;
    off = (off + bytes + 255) & ~(size_t)255;
    return p;
  };
  __bf16* xbf  = (__bf16*)carve((size_t)NN * DIN * 2);
  __bf16* w0t  = (__bf16*)carve((size_t)FH * DIN * 2);
  __bf16* w1t  = (__bf16*)carve((size_t)HF1 * FH * 2);
  __bf16* r1t  = (__bf16*)carve((size_t)HF1 * FH * 2);
  __bf16* w2t  = (__bf16*)carve((size_t)HF2 * HF1 * 2);
  __bf16* r2t  = (__bf16*)carve((size_t)HF2 * HF1 * 2);
  float*  feat = (float*) carve((size_t)NN * HF1 * 4);
  float*  res  = (float*) carve((size_t)NN * HF1 * 4);
  float*  rst  = (float*) carve((size_t)NN * HF1 * 4);
  float*  hf   = (float*) carve((size_t)NN * HF1 * 4);
  __bf16* hbf  = (__bf16*)carve((size_t)NN * HF1 * 2);
  float*  el   = (float*) carve((size_t)NN * HH * 4);
  float*  er   = (float*) carve((size_t)NN * HH * 4);
  float*  ebuf = (float*) carve((size_t)EE * HH * 4);
  int*    mbuf = (int*)   carve((size_t)NN * HH * 4);
  float*  den  = (float*) carve((size_t)NN * HH * 4);
  float*  alp  = (float*) carve((size_t)EE * HH * 4);

  const int BS = 256;
  auto g1 = [](long n) { return dim3((unsigned)((n + 255) / 256)); };
  const dim3 gblk(32, 4);

  // ---- prep: bf16 activations + transposed bf16 weights
  k_cvt_bf16      <<<g1((long)NN * DIN), BS, 0, stream>>>(xbf, x, (long)NN * DIN);
  k_transpose_bf16<<<g1((long)DIN * FH),  BS, 0, stream>>>(w0t, W0,    DIN, FH);
  k_transpose_bf16<<<g1((long)FH * HF1),  BS, 0, stream>>>(w1t, W1,    FH,  HF1);
  k_transpose_bf16<<<g1((long)FH * HF1),  BS, 0, stream>>>(r1t, resW1, FH,  HF1);
  k_transpose_bf16<<<g1((long)HF1 * HF2), BS, 0, stream>>>(w2t, W2,    HF1, HF2);
  k_transpose_bf16<<<g1((long)HF1 * HF2), BS, 0, stream>>>(r2t, resW2, HF1, HF2);

  // =================== layer 0: H=1, F=128, K=512, no residual, ELU
  k_gemm_bf16_wmma<<<dim3(NN / 16, FH / 64), gblk, 0, stream>>>(xbf, w0t, feat, FH, DIN);
  k_dots<<<g1((long)NN * 1), BS, 0, stream>>>(feat, al0, ar0, el, er, NN, 1, FH);
  k_fill_i32<<<g1((long)NN * 1),  BS, 0, stream>>>(mbuf, (int)0x80000000, (long)NN * 1);
  k_fill_f32<<<g1((long)NN * 1),  BS, 0, stream>>>(den, 0.f, (long)NN * 1);
  k_fill_f32<<<g1((long)NN * FH), BS, 0, stream>>>(rst, 0.f, (long)NN * FH);
  k_edge_e    <<<g1((long)EE * 1), BS, 0, stream>>>(src, dst, el, er, ebuf, mbuf, EE, 1);
  k_edge_exp  <<<g1((long)EE * 1), BS, 0, stream>>>(dst, ebuf, mbuf, den, EE, 1);
  k_edge_alpha<<<g1((long)EE * 1), BS, 0, stream>>>(dst, ebuf, den, alp, EE, 1);
  k_edge_msg  <<<g1((long)EE * 1 * (FH / 4)), BS, 0, stream>>>(src, dst, alp, feat, rst, EE, 1, FH);
  k_post<<<g1((long)NN * FH), BS, 0, stream>>>(rst, (const float*)nullptr, b0, hf, hbf, FH, 1, (long)NN * FH);

  // =================== layer 1: H=4, F=128, K=128, residual, ELU
  k_gemm_bf16_wmma<<<dim3(NN / 16, HF1 / 64), gblk, 0, stream>>>(hbf, w1t, feat, HF1, FH);
  k_gemm_bf16_wmma<<<dim3(NN / 16, HF1 / 64), gblk, 0, stream>>>(hbf, r1t, res,  HF1, FH);
  k_dots<<<g1((long)NN * HH), BS, 0, stream>>>(feat, al1, ar1, el, er, NN, HH, FH);
  k_fill_i32<<<g1((long)NN * HH),  BS, 0, stream>>>(mbuf, (int)0x80000000, (long)NN * HH);
  k_fill_f32<<<g1((long)NN * HH),  BS, 0, stream>>>(den, 0.f, (long)NN * HH);
  k_fill_f32<<<g1((long)NN * HF1), BS, 0, stream>>>(rst, 0.f, (long)NN * HF1);
  k_edge_e    <<<g1((long)EE * HH), BS, 0, stream>>>(src, dst, el, er, ebuf, mbuf, EE, HH);
  k_edge_exp  <<<g1((long)EE * HH), BS, 0, stream>>>(dst, ebuf, mbuf, den, EE, HH);
  k_edge_alpha<<<g1((long)EE * HH), BS, 0, stream>>>(dst, ebuf, den, alp, EE, HH);
  k_edge_msg  <<<g1((long)EE * HH * (FH / 4)), BS, 0, stream>>>(src, dst, alp, feat, rst, EE, HH, FH);
  k_post<<<g1((long)NN * HF1), BS, 0, stream>>>(rst, res, b1, hf, hbf, HF1, 1, (long)NN * HF1);
  k_prior<<<g1((long)NN), BS, 0, stream>>>(hf, prior, NN, HF1);

  // =================== layer 2: H=4, F=64, K=512, residual, no activation
  k_gemm_bf16_wmma<<<dim3(NN / 16, HF2 / 64), gblk, 0, stream>>>(hbf, w2t, feat, HF2, HF1);
  k_gemm_bf16_wmma<<<dim3(NN / 16, HF2 / 64), gblk, 0, stream>>>(hbf, r2t, res,  HF2, HF1);
  k_dots<<<g1((long)NN * HH), BS, 0, stream>>>(feat, al2, ar2, el, er, NN, HH, CC);
  k_fill_i32<<<g1((long)NN * HH),  BS, 0, stream>>>(mbuf, (int)0x80000000, (long)NN * HH);
  k_fill_f32<<<g1((long)NN * HH),  BS, 0, stream>>>(den, 0.f, (long)NN * HH);
  k_fill_f32<<<g1((long)NN * HF2), BS, 0, stream>>>(rst, 0.f, (long)NN * HF2);
  k_edge_e    <<<g1((long)EE * HH), BS, 0, stream>>>(src, dst, el, er, ebuf, mbuf, EE, HH);
  k_edge_exp  <<<g1((long)EE * HH), BS, 0, stream>>>(dst, ebuf, mbuf, den, EE, HH);
  k_edge_alpha<<<g1((long)EE * HH), BS, 0, stream>>>(dst, ebuf, den, attout, EE, HH); // att output
  k_edge_msg  <<<g1((long)EE * HH * (CC / 4)), BS, 0, stream>>>(src, dst, attout, feat, rst, EE, HH, CC);
  k_final<<<g1((long)NN * CC), BS, 0, stream>>>(rst, res, b2, logits, NN, HH, CC);

  (void)in_sizes; (void)n_in; (void)out_size; (void)ws_size;
}
